// DeformableAttention_17351667875863
// MI455X (gfx1250) — matrix operand order repaired and checked by reading, not compiled, not run on hardware
//
#include <hip/hip_runtime.h>
#include <hip/hip_bf16.h>
#include <math.h>

typedef __bf16 bf16_t;
typedef bf16_t v16bf __attribute__((ext_vector_type(16)));
typedef float  v8f   __attribute__((ext_vector_type(8)));
typedef unsigned int v8u __attribute__((ext_vector_type(8)));
typedef unsigned int u32x4 __attribute__((ext_vector_type(4)));
typedef int i32x8 __attribute__((ext_vector_type(8)));
typedef int i32x4 __attribute__((ext_vector_type(4)));

#define DEVI __device__ __forceinline__

DEVI unsigned short f2bf(float f) {
  unsigned int u = __float_as_uint(f);
  u += 0x7FFFu + ((u >> 16) & 1u);          // round-to-nearest-even
  return (unsigned short)(u >> 16);
}

// A-fragment K-pair offset for 16x16x32 bf16 WMMA (contiguous k = 16*hl + s, matches B fills)
DEVI int kofsA(int v, int hl) { return 16 * hl + 2 * v; }

// ---- DPP16 butterfly reductions over each 16-lane row (VALU only) ----
// quad_perm(1,0,3,2)=0xB1, quad_perm(2,3,0,1)=0x4E, row_ror:4=0x124, row_ror:8=0x128
DEVI float red_max16(float x) {
  int t;
  t = __builtin_amdgcn_update_dpp(0, __float_as_int(x), 0xB1, 0xF, 0xF, true);
  x = fmaxf(x, __int_as_float(t));
  t = __builtin_amdgcn_update_dpp(0, __float_as_int(x), 0x4E, 0xF, 0xF, true);
  x = fmaxf(x, __int_as_float(t));
  t = __builtin_amdgcn_update_dpp(0, __float_as_int(x), 0x124, 0xF, 0xF, true);
  x = fmaxf(x, __int_as_float(t));
  t = __builtin_amdgcn_update_dpp(0, __float_as_int(x), 0x128, 0xF, 0xF, true);
  x = fmaxf(x, __int_as_float(t));
  return x;
}
DEVI float red_sum16(float x) {
  int t;
  t = __builtin_amdgcn_update_dpp(0, __float_as_int(x), 0xB1, 0xF, 0xF, true);
  x += __int_as_float(t);
  t = __builtin_amdgcn_update_dpp(0, __float_as_int(x), 0x4E, 0xF, 0xF, true);
  x += __int_as_float(t);
  t = __builtin_amdgcn_update_dpp(0, __float_as_int(x), 0x124, 0xF, 0xF, true);
  x += __int_as_float(t);
  t = __builtin_amdgcn_update_dpp(0, __float_as_int(x), 0x128, 0xF, 0xF, true);
  x += __int_as_float(t);
  return x;
}

// ---- Tensor Data Mover: 2-D tile load (f32) global -> LDS ----
// Loads tile_dim0=32 cols x tile_dim1=128 rows from a [*,1024]-strided f32
// tensor into LDS (packed rows of 128B). D# packing per ISA 8.3/8.4.
DEVI void tdm_load_bias_tile(const float* gptr, unsigned lds_off) {
  unsigned long long ga = (unsigned long long)(const void*)gptr;
  u32x4 g0;
  g0[0] = 1u;                                            // count=1 (user D#)
  g0[1] = lds_off;                                       // lds_addr (bytes)
  g0[2] = (unsigned)(ga & 0xFFFFFFFFull);                // global_addr[31:0]
  g0[3] = (unsigned)((ga >> 32) & 0x1FFFFFFull) | 0x80000000u;  // [56:32] | type=2
  i32x8 g1;
  g1[0] = (int)(2u << 16);        // workgroup_mask=0, data_size=2 (4 bytes)
  g1[1] = (int)(1024u << 16);     // tensor_dim0[15:0]=1024 in [63:48]
  g1[2] = (int)(1024u << 16);     // tensor_dim0 hi=0 | tensor_dim1[15:0]=1024
  g1[3] = (int)(32u << 16);       // tensor_dim1 hi=0 | tile_dim0=32
  g1[4] = 128;                    // tile_dim1=128, tile_dim2=0
  g1[5] = 1024;                   // tensor_dim0_stride[31:0]=1024
  g1[6] = 0;                      // stride hi | tensor_dim1_stride lo
  g1[7] = 0;
  i32x4 z4;
  z4[0] = 0; z4[1] = 0; z4[2] = 0; z4[3] = 0;
  i32x8 z8;
#pragma unroll
  for (int i = 0; i < 8; i++) z8[i] = 0;
  // 6-arg form (clang-23 / therock headers): g0, g1, g2, g3, extra, cpol
  __builtin_amdgcn_tensor_load_to_lds(g0, g1, z4, z4, z8, 0);
}

// ---------------------------------------------------------------------------
// f32 -> bf16 conversion
// ---------------------------------------------------------------------------
__global__ void cvt_f32_bf16(const float* __restrict__ src,
                             unsigned short* __restrict__ dst, int n) {
  int i = blockIdx.x * blockDim.x + threadIdx.x;
  if (i < n) dst[i] = f2bf(src[i]);
}

// ---------------------------------------------------------------------------
// Generic WMMA GEMM: Y[M,N] = Abf[M,K] * Bbf[N,K]^T + bias[N]
// MODE 0: f32 row-major output.
// MODE 1: Q path: scale by 0.125 and repack bf16 to [b,h,n,d].
// block = 128 threads (4 waves); wave -> 16x64 tile; grid = (N/64, M/64)
// ---------------------------------------------------------------------------
template <int MODE>
__global__ void gemm_bf16_wmma(const unsigned short* __restrict__ A,
                               const unsigned short* __restrict__ B,
                               const float* __restrict__ bias,
                               void* __restrict__ out,
                               int M, int N, int K) {
  const int wave  = threadIdx.x >> 5;
  const int lane  = threadIdx.x & 31;
  const int lanen = lane & 15;
  const int hl    = lane >> 4;
  const int m0 = blockIdx.y * 64 + wave * 16;
  const int n0 = blockIdx.x * 64;
  (void)M;

  v8f acc[4];
#pragma unroll
  for (int t = 0; t < 4; t++)
#pragma unroll
    for (int v = 0; v < 8; v++) acc[t][v] = 0.f;

  const unsigned int* Arow = (const unsigned int*)(A + (size_t)(m0 + lanen) * K);
  for (int k0 = 0; k0 < K; k0 += 32) {
    v8u au;
#pragma unroll
    for (int v = 0; v < 8; v++) au[v] = Arow[(k0 + kofsA(v, hl)) >> 1];
    v16bf av = __builtin_bit_cast(v16bf, au);
#pragma unroll
    for (int t = 0; t < 4; t++) {
      const unsigned int* Brow =
          (const unsigned int*)(B + (size_t)(n0 + t * 16 + lanen) * K);
      v8u bu;
#pragma unroll
      for (int v = 0; v < 8; v++) bu[v] = Brow[(k0 + 16 * hl + 2 * v) >> 1];
      v16bf bv = __builtin_bit_cast(v16bf, bu);
      acc[t] = __builtin_amdgcn_wmma_f32_16x16x32_bf16(
          false, av, false, bv, (short)0, acc[t], false, false);
    }
  }

#pragma unroll
  for (int t = 0; t < 4; t++)
#pragma unroll
    for (int v = 0; v < 8; v++) {
      int row = m0 + v + 8 * hl;
      int col = n0 + t * 16 + lanen;
      float y = acc[t][v] + bias[col];
      if (MODE == 0) {
        ((float*)out)[(size_t)row * N + col] = y;
      } else {
        int bb = row >> 10, nn = row & 1023, hh = col >> 6, dd = col & 63;
        ((unsigned short*)out)[((((size_t)bb * 8 + hh) * 1024 + nn) << 6) + dd] =
            f2bf(y * 0.125f);
      }
    }
}

// ---------------------------------------------------------------------------
// Depthwise 5x5 conv (zero pad) + bias + exact GELU.  x layout [(b,n),c].
// ---------------------------------------------------------------------------
__global__ void dwconv_gelu(const float* __restrict__ x,
                            const float* __restrict__ w,
                            const float* __restrict__ bb,
                            float* __restrict__ out) {
  int i = blockIdx.x * blockDim.x + threadIdx.x;  // 4*1024*512
  if (i >= 4 * 1024 * 512) return;
  int c = i & 511;
  int n = (i >> 9) & 1023;
  int b = i >> 19;
  int yy = n >> 5, xx = n & 31;
  float acc = bb[c];
#pragma unroll
  for (int dy = 0; dy < 5; dy++) {
    int sy = yy + dy - 2;
    if (sy < 0 || sy > 31) continue;
#pragma unroll
    for (int dx = 0; dx < 5; dx++) {
      int sx = xx + dx - 2;
      if (sx < 0 || sx > 31) continue;
      acc += x[((size_t)b * 1024 + sy * 32 + sx) * 512 + c] *
             w[c * 25 + dy * 5 + dx];
    }
  }
  out[i] = 0.5f * acc * (1.f + erff(acc * 0.70710678118654752f));
}

// ---------------------------------------------------------------------------
// Offset head: 1x1 conv to 16ch, tanh, *2.  Off layout [(b,n), 16]
// ---------------------------------------------------------------------------
__global__ void offset_net(const float* __restrict__ hbuf,
                           const float* __restrict__ ow,
                           float* __restrict__ off) {
  int i = blockIdx.x * blockDim.x + threadIdx.x;  // 4*1024*16
  if (i >= 4 * 1024 * 16) return;
  int oc = i & 15;
  int row = i >> 4;
  const float* hp = hbuf + (size_t)row * 512;
  const float* wp = ow + (size_t)oc * 512;
  float acc = 0.f;
  for (int c = 0; c < 512; c++) acc += hp[c] * wp[c];
  off[i] = tanhf(acc) * 2.f;
}

// ---------------------------------------------------------------------------
// Bilinear grid sample (zeros padding, align_corners=False) of K and V.
// Writes Ks as bf16 [bh, n, d] and Vs transposed as bf16 [bh, d, n].
// ---------------------------------------------------------------------------
__global__ void grid_sample_kv(const float* __restrict__ Kf,
                               const float* __restrict__ Vf,
                               const float* __restrict__ off,
                               unsigned short* __restrict__ Ks,
                               unsigned short* __restrict__ Vst) {
  int i = blockIdx.x * blockDim.x + threadIdx.x;  // 32*1024*64
  if (i >= 32 * 1024 * 64) return;
  int d = i & 63;
  int n = (i >> 6) & 1023;
  int bh = i >> 16;
  int b = bh >> 3, h = bh & 7;
  int yy = n >> 5, xx = n & 31;
  float gx = -1.f + (2.f / 31.f) * (float)xx;
  float gy = -1.f + (2.f / 31.f) * (float)yy;
  float sx = gx + off[((size_t)b * 1024 + n) * 16 + 2 * h];
  float sy = gy + off[((size_t)b * 1024 + n) * 16 + 2 * h + 1];
  sx = fminf(fmaxf(sx, -1.f), 1.f);
  sy = fminf(fmaxf(sy, -1.f), 1.f);
  float fx = (sx + 1.f) * 16.f - 0.5f;
  float fy = (sy + 1.f) * 16.f - 0.5f;
  float x0 = floorf(fx), y0 = floorf(fy);
  float ak = 0.f, av = 0.f;
#pragma unroll
  for (int dy = 0; dy < 2; dy++)
#pragma unroll
    for (int dx = 0; dx < 2; dx++) {
      float xi = x0 + dx, yi = y0 + dy;
      float w = (1.f - fabsf(fx - xi)) * (1.f - fabsf(fy - yi));
      if (xi < 0.f || xi > 31.f || yi < 0.f || yi > 31.f) w = 0.f;
      int xic = (int)fminf(fmaxf(xi, 0.f), 31.f);
      int yic = (int)fminf(fmaxf(yi, 0.f), 31.f);
      size_t src = ((size_t)b * 1024 + yic * 32 + xic) * 512 + h * 64 + d;
      ak += Kf[src] * w;
      av += Vf[src] * w;
    }
  Ks[((size_t)bh * 1024 + n) * 64 + d] = f2bf(ak);
  Vst[((size_t)bh * 64 + d) * 1024 + n] = f2bf(av);
}

// ---------------------------------------------------------------------------
// Relative-position bias: gather table -> bias0 [64,64]
// ---------------------------------------------------------------------------
__global__ void bias_gather(const float* __restrict__ table,
                            float* __restrict__ bias0) {
  int i = blockIdx.x * blockDim.x + threadIdx.x;  // 4096
  if (i >= 4096) return;
  int p = i >> 6, q = i & 63;
  int py = p >> 3, px = p & 7, qy = q >> 3, qx = q & 7;
  int idx = (py - qy + 7) * 15 + (px - qx + 7);
  bias0[i] = table[idx];
}

// Bicubic (a=-0.75, half-pixel, border-replicate) resize matrix rows [1024,64]
__global__ void bicubic_rows(float* __restrict__ Mrow) {
  int o = blockIdx.x * blockDim.x + threadIdx.x;
  if (o >= 1024) return;
  float* r = Mrow + (size_t)o * 64;
  for (int j = 0; j < 64; j++) r[j] = 0.f;
  const float a = -0.75f;
  float src = ((float)o + 0.5f) * (64.f / 1024.f) - 0.5f;
  float i0 = floorf(src);
  float t = src - i0;
  for (int k = -1; k <= 2; k++) {
    int idx = (int)i0 + k;
    idx = idx < 0 ? 0 : (idx > 63 ? 63 : idx);
    float dd = fabsf(t - (float)k);
    float w;
    if (dd <= 1.f)      w = ((a + 2.f) * dd - (a + 3.f)) * dd * dd + 1.f;
    else if (dd < 2.f)  w = (((a * dd - 5.f * a) * dd + 8.f * a) * dd - 4.f * a);
    else                w = 0.f;
    r[idx] += w;
  }
}

// small f32 matmul (bias path): C[M,N] = A[M,K] * (transB ? B[N,K]^T : B[K,N])
__global__ void matmul_small(const float* __restrict__ A,
                             const float* __restrict__ B,
                             float* __restrict__ C,
                             int M, int N, int K, int transB) {
  int i = blockIdx.x * blockDim.x + threadIdx.x;
  if (i >= M * N) return;
  int m = i / N, n = i % N;
  float acc = 0.f;
  if (transB)
    for (int k = 0; k < K; k++) acc += A[(size_t)m * K + k] * B[(size_t)n * K + k];
  else
    for (int k = 0; k < K; k++) acc += A[(size_t)m * K + k] * B[(size_t)k * N + n];
  C[i] = acc;
}

// ---------------------------------------------------------------------------
// Flash attention: per wave 16 query rows of one (b,h); online softmax over
// 32-key blocks. S and O via v_wmma_f32_16x16x32_bf16; P converted to a bf16
// A-fragment through a per-wave LDS tile. Q is pre-scaled by 1/sqrt(64).
// Bias tiles (128 q-rows x 32 keys, f32) are streamed into LDS by the Tensor
// Data Mover, double-buffered; wave 0 issues TDM + s_wait_tensorcnt, block
// barrier hands the tile to all 8 waves.
// block = 256 threads (8 waves) -> 128 query rows; grid = (32, 8).
// ---------------------------------------------------------------------------
__global__ void flash_attn(const unsigned short* __restrict__ Qbf,
                           const unsigned short* __restrict__ Ksbf,
                           const unsigned short* __restrict__ Vstbf,
                           const float* __restrict__ biasMat,
                           unsigned short* __restrict__ Obf) {
  __shared__ unsigned short ldsP[8 * 512];   // 8 waves * (16x32 bf16)
  __shared__ float biasT[2][128 * 32];       // double-buffered bias tiles (TDM)
  const int wave  = threadIdx.x >> 5;
  const int lane  = threadIdx.x & 31;
  const int lanen = lane & 15;
  const int hl    = lane >> 4;
  const int bh = blockIdx.x;
  const int b = bh >> 3, h = bh & 7;
  const int q0 = blockIdx.y * 128 + wave * 16;

  const float* biasBase = biasMat + (size_t)(blockIdx.y * 128) * 1024;
  unsigned ldsBias0 = (unsigned)(uintptr_t)(&biasT[0][0]);
  unsigned ldsBias1 = (unsigned)(uintptr_t)(&biasT[1][0]);

  // preload bias tile for j0 = 0
  if (wave == 0) {
    tdm_load_bias_tile(biasBase, ldsBias0);
    __builtin_amdgcn_s_wait_tensorcnt(0);
  }
  __syncthreads();

  // preload Q A-fragments (d = 64 -> 2 fragments)
  const unsigned int* Qrow =
      (const unsigned int*)(Qbf + ((size_t)bh * 1024 + q0 + lanen) * 64);
  v16bf aq[2];
#pragma unroll
  for (int kk = 0; kk < 2; kk++) {
    v8u u;
#pragma unroll
    for (int v = 0; v < 8; v++) u[v] = Qrow[(kk * 32 + kofsA(v, hl)) >> 1];
    aq[kk] = __builtin_bit_cast(v16bf, u);
  }

  v8f o[4];
  float mrun[8], lrun[8];
#pragma unroll
  for (int f = 0; f < 4; f++)
#pragma unroll
    for (int v = 0; v < 8; v++) o[f][v] = 0.f;
#pragma unroll
  for (int v = 0; v < 8; v++) { mrun[v] = -1e30f; lrun[v] = 0.f; }

  unsigned short* pl = ldsP + wave * 512;

  for (int j0 = 0; j0 < 1024; j0 += 32) {
    const int cur = (j0 >> 5) & 1;
    // kick off TDM for the next bias tile into the other buffer
    if (wave == 0 && (j0 + 32) < 1024)
      tdm_load_bias_tile(biasBase + (j0 + 32), cur ? ldsBias0 : ldsBias1);

    // ---- S = Q * K^T (two 16x16 tiles) ----
    v8f s[2];
#pragma unroll
    for (int t = 0; t < 2; t++) {
      v8f z;
#pragma unroll
      for (int v = 0; v < 8; v++) z[v] = 0.f;
      const unsigned int* Krow = (const unsigned int*)(
          Ksbf + ((size_t)bh * 1024 + j0 + t * 16 + lanen) * 64);
#pragma unroll
      for (int kk = 0; kk < 2; kk++) {
        v8u bu;
#pragma unroll
        for (int v = 0; v < 8; v++) bu[v] = Krow[(kk * 32 + 16 * hl + 2 * v) >> 1];
        v16bf bv = __builtin_bit_cast(v16bf, bu);
        z = __builtin_amdgcn_wmma_f32_16x16x32_bf16(
            false, aq[kk], false, bv, (short)0, z, false, false);
      }
      s[t] = z;
    }

    // ---- add relative-position bias from the LDS tile ----
    const float* bt = &biasT[cur][0];
#pragma unroll
    for (int t = 0; t < 2; t++)
#pragma unroll
      for (int v = 0; v < 8; v++)
        s[t][v] += bt[(wave * 16 + v + 8 * hl) * 32 + t * 16 + lanen];

    // ---- online softmax stats (DPP row reductions, VALU only) ----
    float alpha[8], rs[8];
#pragma unroll
    for (int v = 0; v < 8; v++) {
      float mt = red_max16(fmaxf(s[0][v], s[1][v]));
      float mn = fmaxf(mrun[v], mt);
      alpha[v] = __expf(mrun[v] - mn);
      mrun[v] = mn;
    }
#pragma unroll
    for (int v = 0; v < 8; v++) {
      float p0 = __expf(s[0][v] - mrun[v]);
      float p1 = __expf(s[1][v] - mrun[v]);
      s[0][v] = p0;
      s[1][v] = p1;
      rs[v] = red_sum16(p0 + p1);
    }
#pragma unroll
    for (int v = 0; v < 8; v++) lrun[v] = lrun[v] * alpha[v] + rs[v];
#pragma unroll
    for (int f = 0; f < 4; f++)
#pragma unroll
      for (int v = 0; v < 8; v++) o[f][v] *= alpha[v];

    // ---- P (C-layout) -> LDS -> bf16 A-fragment ----
#pragma unroll
    for (int t = 0; t < 2; t++)
#pragma unroll
      for (int v = 0; v < 8; v++)
        pl[(v + 8 * hl) * 32 + t * 16 + lanen] = f2bf(s[t][v]);

    v8u pu;
#pragma unroll
    for (int v = 0; v < 8; v++) {
      int k = kofsA(v, hl);
      unsigned int lo = pl[lanen * 32 + k];
      unsigned int hi = pl[lanen * 32 + k + 1];
      pu[v] = lo | (hi << 16);
    }
    v16bf ap = __builtin_bit_cast(v16bf, pu);

    // ---- O += P * V  (V stored transposed [d, n] -> B^T pattern) ----
#pragma unroll
    for (int f = 0; f < 4; f++) {
      const unsigned int* Vrow = (const unsigned int*)(
          Vstbf + ((size_t)bh * 64 + f * 16 + lanen) * 1024 + j0);
      v8u bu;
#pragma unroll
      for (int v = 0; v < 8; v++) bu[v] = Vrow[(16 * hl + 2 * v) >> 1];
      v16bf bv = __builtin_bit_cast(v16bf, bu);
      o[f] = __builtin_amdgcn_wmma_f32_16x16x32_bf16(
          false, ap, false, bv, (short)0, o[f], false, false);
    }

    // hand next bias tile to all waves
    if (wave == 0) __builtin_amdgcn_s_wait_tensorcnt(0);
    __syncthreads();
  }

  // ---- normalize and write attn output as bf16 [(b,n), h*64+d] ----
#pragma unroll
  for (int f = 0; f < 4; f++)
#pragma unroll
    for (int v = 0; v < 8; v++) {
      int qr = q0 + v + 8 * hl;
      int dc = h * 64 + f * 16 + lanen;
      Obf[((size_t)b * 1024 + qr) * 512 + dc] = f2bf(o[f][v] / lrun[v]);
    }
}

// ---------------------------------------------------------------------------
extern "C" void kernel_launch(void* const* d_in, const int* in_sizes, int n_in,
                              void* d_out, int out_size, void* d_ws, size_t ws_size,
                              hipStream_t stream) {
  (void)in_sizes; (void)n_in; (void)out_size; (void)ws_size;

  const float* x    = (const float*)d_in[0];
  const float* q_w  = (const float*)d_in[1];
  const float* q_b  = (const float*)d_in[2];
  const float* k_w  = (const float*)d_in[3];
  const float* k_b  = (const float*)d_in[4];
  const float* v_w  = (const float*)d_in[5];
  const float* v_b  = (const float*)d_in[6];
  const float* o_w  = (const float*)d_in[7];
  const float* o_b  = (const float*)d_in[8];
  const float* dw_w = (const float*)d_in[9];
  const float* dw_b = (const float*)d_in[10];
  const float* offw = (const float*)d_in[11];
  const float* btab = (const float*)d_in[12];

  size_t off = 0;
  auto alloc = [&](size_t bytes) -> char* {
    char* p = (char*)d_ws + off;
    off += (bytes + 255) & ~(size_t)255;
    return p;
  };

  unsigned short* Xbf   = (unsigned short*)alloc((size_t)4096 * 512 * 2);
  unsigned short* Wqbf  = (unsigned short*)alloc((size_t)512 * 512 * 2);
  unsigned short* Wkbf  = (unsigned short*)alloc((size_t)512 * 512 * 2);
  unsigned short* Wvbf  = (unsigned short*)alloc((size_t)512 * 512 * 2);
  unsigned short* Wobf  = (unsigned short*)alloc((size_t)512 * 512 * 2);
  unsigned short* Qbf   = (unsigned short*)alloc((size_t)32 * 1024 * 64 * 2);
  float*          Kf    = (float*)alloc((size_t)4096 * 512 * 4);
  float*          Vf    = (float*)alloc((size_t)4096 * 512 * 4);
  float*          Hf    = (float*)alloc((size_t)4096 * 512 * 4);
  float*          Off   = (float*)alloc((size_t)4096 * 16 * 4);
  unsigned short* Ksbf  = (unsigned short*)alloc((size_t)32 * 1024 * 64 * 2);
  unsigned short* Vstbf = (unsigned short*)alloc((size_t)32 * 64 * 1024 * 2);
  float*          bias0 = (float*)alloc((size_t)64 * 64 * 4);
  float*          Mmat  = (float*)alloc((size_t)1024 * 64 * 4);
  float*          Tmat  = (float*)alloc((size_t)1024 * 64 * 4);
  float*          biasM = (float*)alloc((size_t)1024 * 1024 * 4);
  unsigned short* Attn  = (unsigned short*)alloc((size_t)4096 * 512 * 2);

  // --- bf16 conversions ---
  cvt_f32_bf16<<<(4096 * 512 + 255) / 256, 256, 0, stream>>>(x, Xbf, 4096 * 512);
  cvt_f32_bf16<<<(512 * 512 + 255) / 256, 256, 0, stream>>>(q_w, Wqbf, 512 * 512);
  cvt_f32_bf16<<<(512 * 512 + 255) / 256, 256, 0, stream>>>(k_w, Wkbf, 512 * 512);
  cvt_f32_bf16<<<(512 * 512 + 255) / 256, 256, 0, stream>>>(v_w, Wvbf, 512 * 512);
  cvt_f32_bf16<<<(512 * 512 + 255) / 256, 256, 0, stream>>>(o_w, Wobf, 512 * 512);

  // --- projections (WMMA) ---
  dim3 gblk(128);
  dim3 ggrid(512 / 64, 4096 / 64);
  gemm_bf16_wmma<1><<<ggrid, gblk, 0, stream>>>(Xbf, Wqbf, q_b, (void*)Qbf, 4096, 512, 512);
  gemm_bf16_wmma<0><<<ggrid, gblk, 0, stream>>>(Xbf, Wkbf, k_b, (void*)Kf,  4096, 512, 512);
  gemm_bf16_wmma<0><<<ggrid, gblk, 0, stream>>>(Xbf, Wvbf, v_b, (void*)Vf,  4096, 512, 512);

  // --- offset network + deformable sampling ---
  dwconv_gelu<<<(4 * 1024 * 512 + 255) / 256, 256, 0, stream>>>(x, dw_w, dw_b, Hf);
  offset_net<<<(4 * 1024 * 16 + 255) / 256, 256, 0, stream>>>(Hf, offw, Off);
  grid_sample_kv<<<(32 * 1024 * 64 + 255) / 256, 256, 0, stream>>>(Kf, Vf, Off, Ksbf, Vstbf);

  // --- relative position bias: bias = M @ bias0 @ M^T ---
  bias_gather<<<16, 256, 0, stream>>>(btab, bias0);
  bicubic_rows<<<4, 256, 0, stream>>>(Mmat);
  matmul_small<<<(1024 * 64 + 255) / 256, 256, 0, stream>>>(Mmat, bias0, Tmat, 1024, 64, 64, 0);
  matmul_small<<<(1024 * 1024 + 255) / 256, 256, 0, stream>>>(Tmat, Mmat, biasM, 1024, 1024, 64, 1);

  // --- flash attention (WMMA + TDM) ---
  flash_attn<<<dim3(32, 8), 256, 0, stream>>>(Qbf, Ksbf, Vstbf, biasM, Attn);

  // --- output projection (WMMA) -> d_out [B,N,C] f32 ---
  gemm_bf16_wmma<0><<<ggrid, gblk, 0, stream>>>(Attn, Wobf, o_b, d_out, 4096, 512, 512);
}
